// ProbSparseAttention_16003048145404
// MI455X (gfx1250) — matrix operand order repaired and checked by reading, not compiled, not run on hardware
//
#include <hip/hip_runtime.h>
#include <hip/hip_bf16.h>
#include <math.h>
#include <stdint.h>

// ---------------------------------------------------------------------------
// ProbSparse attention, MI455X (gfx1250, wave32).
// B=4, L=4096, D=512, H=8, DK=64, NTOP=45.
// Heavy GEMMs (4 projections, 34 GFLOP) via v_wmma_f32_16x16x32_bf16,
// 32x32 output tile per wave (4 WMMAs / K-step, 2x fragment reuse).
// ---------------------------------------------------------------------------

#define BB 4
#define LL 4096
#define DD 512
#define HH 8
#define DK 64
#define NTOP 45

typedef __attribute__((ext_vector_type(16))) __bf16 v16bf;
typedef __attribute__((ext_vector_type(8)))  float  v8f;

__device__ __forceinline__ uint32_t pack2bf(float a, float b) {
    __bf16 x = (__bf16)a, y = (__bf16)b;
    uint16_t xi = __builtin_bit_cast(uint16_t, x);
    uint16_t yi = __builtin_bit_cast(uint16_t, y);
    return (uint32_t)xi | ((uint32_t)yi << 16);
}

// ------------------------------- projections -------------------------------
// C[M x N] = A[M x K] * W[K x N] + bias ;  M = B*L = 16384, K = N = 512.
// mode 0: A is row-major (B*L, D).
// mode 1: A is ctx laid out (B, H, L, DK): A[m][k] = ctx[b][h][l][dk],
//         m = b*L + l, k = h*DK + dk.
__device__ __forceinline__ float loadA(const float* __restrict__ A, int m, int k, int mode) {
    if (mode == 0) return A[(size_t)m * DD + k];
    int b = m >> 12, l = m & (LL - 1);
    int h = k >> 6,  dk = k & (DK - 1);
    return A[((((size_t)b * HH + h) * LL) + l) * DK + dk];
}

__global__ __launch_bounds__(256) void proj_wmma(const float* __restrict__ A,
                                                 const float* __restrict__ W,
                                                 const float* __restrict__ bias,
                                                 float* __restrict__ C,
                                                 int mode) {
    // Block tile: 64 rows (M) x 128 cols (N). 8 waves in 2x4 grid; each wave
    // computes a 32x32 tile = 2x2 WMMA subtiles (4 WMMAs / K-step).
    __shared__ __align__(16) __bf16 lA[64][34];    // [m][k], padded rows
    __shared__ __align__(16) __bf16 lB[128][34];   // [n][k] (W transposed)

    const int m0 = blockIdx.x * 64;
    const int n0 = blockIdx.y * 128;

    const int tid  = threadIdx.x;
    const int lane = tid & 31;
    const int wave = tid >> 5;        // 0..7
    const int wm   = wave >> 2;       // 0..1  (32-row band)
    const int wn   = wave & 3;        // 0..3  (32-col band)

    v8f acc[2][2];
    #pragma unroll
    for (int i = 0; i < 2; ++i)
        #pragma unroll
        for (int j = 0; j < 2; ++j)
            acc[i][j] = (v8f){0.f, 0.f, 0.f, 0.f, 0.f, 0.f, 0.f, 0.f};

    // staging coordinates
    const int ar  = tid >> 2;          // 0..63   A row
    const int akg = (tid & 3) * 8;     // A k-group (8 contiguous k)
    const int bp  = tid >> 4;          // 0..15   W k-pair index (k = 2*bp, 2*bp+1)
    const int bn8 = (tid & 15) * 8;    // W col group (8 contiguous n)

    for (int k0 = 0; k0 < DD; k0 += 32) {
        // prefetch next W tile (emits global_prefetch_b8)
        if (k0 + 32 < DD)
            __builtin_prefetch(&W[(size_t)(k0 + 32 + 2 * bp) * DD + n0 + bn8], 0, 1);

        __syncthreads();
        // --- stage A (64x32 f32 -> bf16), packed b32 LDS stores -------------
        {
            float f[8];
            #pragma unroll
            for (int j = 0; j < 8; ++j)
                f[j] = loadA(A, m0 + ar, k0 + akg + j, mode);
            #pragma unroll
            for (int j = 0; j < 4; ++j)
                *(uint32_t*)&lA[ar][akg + 2 * j] = pack2bf(f[2 * j], f[2 * j + 1]);
        }
        // --- stage W (32x128 f32 -> bf16 transposed to [n][k]) --------------
        {
            const float* w0 = &W[(size_t)(k0 + 2 * bp) * DD + n0 + bn8];
            const float* w1 = w0 + DD;
            #pragma unroll
            for (int j = 0; j < 8; ++j)
                *(uint32_t*)&lB[bn8 + j][2 * bp] = pack2bf(w0[j], w1[j]);
        }
        __syncthreads();

        // --- fragments (ISA Sec. 7.12.2 16-bit lane layouts) ----------------
        // A 16x32: lane<16 -> row=lane,   k in {0..7,16..23};
        //          lane>=16 -> row=lane-16, k in {8..15,24..31}
        v16bf af[2];
        #pragma unroll
        for (int sm = 0; sm < 2; ++sm) {
            const int mrow  = wm * 32 + sm * 16 + (lane & 15);
            const int koffA = (lane >> 4) * 8;
            #pragma unroll
            for (int v = 0; v < 8; ++v) {
                int k = ((v < 4) ? (2 * v) : (16 + 2 * (v - 4))) + koffA;
                af[sm][2 * v]     = lA[mrow][k];
                af[sm][2 * v + 1] = lA[mrow][k + 1];
            }
        }
        // B 32x16: lane<16 -> col=lane, k in {0..7,16..23}; lane>=16 -> col=lane-16, k in {8..15,24..31}
        v16bf bfr[2];
        #pragma unroll
        for (int sn = 0; sn < 2; ++sn) {
            const int ncol  = wn * 32 + sn * 16 + (lane & 15);
            const int koffB = (lane >> 4) * 8;
            #pragma unroll
            for (int v = 0; v < 8; ++v) {
                int k = ((v < 4) ? (2 * v) : (16 + 2 * (v - 4))) + koffB;
                bfr[sn][2 * v]     = lB[ncol][k];
                bfr[sn][2 * v + 1] = lB[ncol][k + 1];
            }
        }

        #pragma unroll
        for (int sm = 0; sm < 2; ++sm)
            #pragma unroll
            for (int sn = 0; sn < 2; ++sn)
                acc[sm][sn] = __builtin_amdgcn_wmma_f32_16x16x32_bf16(
                    false, af[sm], false, bfr[sn],
                    (short)0, acc[sm][sn], false, false);
    }

    // C/D layout: VGPR r -> row = r + 8*(lane>=16), col = lane&15
    #pragma unroll
    for (int sm = 0; sm < 2; ++sm) {
        #pragma unroll
        for (int sn = 0; sn < 2; ++sn) {
            const int col   = n0 + wn * 32 + sn * 16 + (lane & 15);
            const int rbase = m0 + wm * 32 + sm * 16 + ((lane >> 4) * 8);
            const float bv  = bias[col];
            #pragma unroll
            for (int r = 0; r < 8; ++r)
                C[(size_t)(rbase + r) * DD + col] = acc[sm][sn][r] + bv;
        }
    }
}

// ------------------------- sparsity measurement M ---------------------------
// One wave32 per (b,h,q): M = max_j dot(Q, K_sample_j) - mean_j dot(...)
__global__ __launch_bounds__(256) void qks_mscore(const float* __restrict__ Qp,
                                                  const float* __restrict__ Kp,
                                                  const int* __restrict__ sidx,
                                                  float* __restrict__ Mout) {
    int gtid = blockIdx.x * blockDim.x + threadIdx.x;
    int wid  = gtid >> 5;
    int lane = gtid & 31;
    int q = wid & (LL - 1);
    int h = (wid >> 12) & (HH - 1);
    int b = wid >> 15;

    const float* qr = Qp + ((size_t)(b * LL + q)) * DD + h * DK;
    float q0 = qr[lane], q1 = qr[lane + 32];

    float mx = -3.4e38f, sm = 0.f;
    for (int j = 0; j < NTOP; ++j) {
        int sj = sidx[j];
        const float* kr = Kp + ((size_t)(b * LL + sj)) * DD + h * DK;
        float part = q0 * kr[lane] + q1 * kr[lane + 32];
        #pragma unroll
        for (int m = 16; m >= 1; m >>= 1) part += __shfl_xor(part, m, 32);
        mx = fmaxf(mx, part);
        sm += part;
    }
    if (lane == 0) Mout[wid] = mx - sm * (1.0f / (float)NTOP);
}

// ------------------------------- top-k (45) ---------------------------------
// One block per (b,h). Descending order, low-index tie-break (matches lax.top_k).
__global__ __launch_bounds__(256) void topk45(const float* __restrict__ Mscore,
                                              int* __restrict__ topIdx) {
    __shared__ float vals[LL];
    __shared__ float rv[256];
    __shared__ int   ri[256];
    int bh = blockIdx.x;
    const float* row = Mscore + (size_t)bh * LL;
    for (int i = threadIdx.x; i < LL; i += 256) vals[i] = row[i];
    __syncthreads();

    for (int r = 0; r < NTOP; ++r) {
        float bv = -3.4e38f; int bi = 0x7fffffff;
        for (int i = threadIdx.x; i < LL; i += 256) {
            float v = vals[i];
            if (v > bv || (v == bv && i < bi)) { bv = v; bi = i; }
        }
        rv[threadIdx.x] = bv; ri[threadIdx.x] = bi;
        __syncthreads();
        for (int s = 128; s > 0; s >>= 1) {
            if (threadIdx.x < s) {
                float ov = rv[threadIdx.x + s]; int oi = ri[threadIdx.x + s];
                if (ov > rv[threadIdx.x] ||
                    (ov == rv[threadIdx.x] && oi < ri[threadIdx.x])) {
                    rv[threadIdx.x] = ov; ri[threadIdx.x] = oi;
                }
            }
            __syncthreads();
        }
        if (threadIdx.x == 0) {
            topIdx[bh * NTOP + r] = ri[0];
            vals[ri[0]] = -3.4e38f;
        }
        __syncthreads();
    }
}

// ---------------------- ctx default = mean(V) broadcast ---------------------
__global__ __launch_bounds__(256) void vmean_fill(const float* __restrict__ Vp,
                                                  float* __restrict__ ctx) {
    int bh = blockIdx.x;
    int b = bh >> 3, h = bh & 7;
    __shared__ float part[4][DK];
    __shared__ float mv[DK];
    int d = threadIdx.x & (DK - 1), seg = threadIdx.x >> 6;

    const float* base = Vp + ((size_t)b * LL) * DD + h * DK + d;
    float acc = 0.f;
    for (int k = seg * 1024; k < (seg + 1) * 1024; ++k)
        acc += base[(size_t)k * DD];
    part[seg][d] = acc;
    __syncthreads();
    if (threadIdx.x < DK)
        mv[threadIdx.x] = (part[0][threadIdx.x] + part[1][threadIdx.x] +
                           part[2][threadIdx.x] + part[3][threadIdx.x]) * (1.0f / (float)LL);
    __syncthreads();

    float* crow = ctx + ((size_t)bh * LL) * DK;
    for (int i = threadIdx.x; i < LL * DK; i += 256)
        crow[i] = mv[i & (DK - 1)];
}

// ---------------- scores + softmax + attn out + ctx update ------------------
// One block per (b,h,u). Scores row resident in LDS.
__global__ __launch_bounds__(256) void scores_softmax_ctx(const float* __restrict__ Qp,
                                                          const float* __restrict__ Kp,
                                                          const float* __restrict__ Vp,
                                                          const int* __restrict__ topIdx,
                                                          float* __restrict__ attn,
                                                          float* __restrict__ ctx) {
    __shared__ float p[LL];
    __shared__ float qv[DK];
    __shared__ float red[256];
    __shared__ float cacc[4][DK];

    int u  = blockIdx.x % NTOP;
    int bh = blockIdx.x / NTOP;
    int b = bh >> 3, h = bh & 7;
    int q = topIdx[bh * NTOP + u];

    if (threadIdx.x < DK)
        qv[threadIdx.x] = Qp[((size_t)(b * LL + q)) * DD + h * DK + threadIdx.x];
    __syncthreads();

    // scores
    float lmax = -3.4e38f;
    for (int k = threadIdx.x; k < LL; k += 256) {
        const float* kr = Kp + ((size_t)(b * LL + k)) * DD + h * DK;
        float dot = 0.f;
        #pragma unroll
        for (int d = 0; d < DK; ++d) dot += qv[d] * kr[d];
        dot *= 0.125f;   // 1/sqrt(64)
        p[k] = dot;
        lmax = fmaxf(lmax, dot);
    }
    red[threadIdx.x] = lmax; __syncthreads();
    for (int s = 128; s > 0; s >>= 1) {
        if (threadIdx.x < s) red[threadIdx.x] = fmaxf(red[threadIdx.x], red[threadIdx.x + s]);
        __syncthreads();
    }
    float mx = red[0]; __syncthreads();

    float lsum = 0.f;
    for (int k = threadIdx.x; k < LL; k += 256) {
        float e = __expf(p[k] - mx);
        p[k] = e;
        lsum += e;
    }
    red[threadIdx.x] = lsum; __syncthreads();
    for (int s = 128; s > 0; s >>= 1) {
        if (threadIdx.x < s) red[threadIdx.x] += red[threadIdx.x + s];
        __syncthreads();
    }
    float inv = 1.0f / red[0]; __syncthreads();

    float* arow = attn + ((size_t)(bh * NTOP + u)) * LL;
    for (int k = threadIdx.x; k < LL; k += 256) {
        float pr = p[k] * inv;
        p[k] = pr;
        arow[k] = pr;
    }
    __syncthreads();

    // ctx row = attn . V
    int d = threadIdx.x & (DK - 1), seg = threadIdx.x >> 6;
    const float* vb = Vp + ((size_t)(b * LL)) * DD + h * DK + d;
    float acc = 0.f;
    for (int k = seg * 1024; k < (seg + 1) * 1024; ++k)
        acc += p[k] * vb[(size_t)k * DD];
    cacc[seg][d] = acc;
    __syncthreads();
    if (threadIdx.x < DK)
        ctx[(((size_t)bh * LL) + q) * DK + threadIdx.x] =
            cacc[0][threadIdx.x] + cacc[1][threadIdx.x] +
            cacc[2][threadIdx.x] + cacc[3][threadIdx.x];
}

// --------------------------------- launch -----------------------------------
extern "C" void kernel_launch(void* const* d_in, const int* in_sizes, int n_in,
                              void* d_out, int out_size, void* d_ws, size_t ws_size,
                              hipStream_t stream) {
    const float* queries = (const float*)d_in[0];
    const float* keys    = (const float*)d_in[1];
    const float* values  = (const float*)d_in[2];
    const int*   sidx    = (const int*)  d_in[3];
    const float* Wq = (const float*)d_in[4];  const float* bq = (const float*)d_in[5];
    const float* Wk = (const float*)d_in[6];  const float* bk = (const float*)d_in[7];
    const float* Wv = (const float*)d_in[8];  const float* bv = (const float*)d_in[9];
    const float* Wo = (const float*)d_in[10]; const float* bo = (const float*)d_in[11];

    const size_t nBLD = (size_t)BB * LL * DD;           // 8,388,608
    char* ws = (char*)d_ws;
    float* Qp     = (float*)(ws);
    float* Kp     = (float*)(ws + nBLD * 4);
    float* Vp     = (float*)(ws + nBLD * 8);
    float* ctx    = (float*)(ws + nBLD * 12);
    float* Mscore = (float*)(ws + nBLD * 16);
    int*   topIdx = (int*)  (ws + nBLD * 16 + (size_t)BB * HH * LL * 4);

    float* out  = (float*)d_out;                 // (B, L, D)
    float* attn = out + nBLD;                    // (B, H, NTOP, L)

    dim3 pgrid(BB * LL / 64, DD / 128);          // 256 x 4 blocks

    // 1) Q/K/V projections (WMMA bf16 GEMMs)
    proj_wmma<<<pgrid, 256, 0, stream>>>(queries, Wq, bq, Qp, 0);
    proj_wmma<<<pgrid, 256, 0, stream>>>(keys,    Wk, bk, Kp, 0);
    proj_wmma<<<pgrid, 256, 0, stream>>>(values,  Wv, bv, Vp, 0);

    // 2) sparsity measure M over sampled keys
    qks_mscore<<<(BB * HH * LL * 32) / 256, 256, 0, stream>>>(Qp, Kp, sidx, Mscore);

    // 3) top-45 queries per (b,h)
    topk45<<<BB * HH, 256, 0, stream>>>(Mscore, topIdx);

    // 4) ctx default = broadcast mean(V)
    vmean_fill<<<BB * HH, 256, 0, stream>>>(Vp, ctx);

    // 5) scores + softmax + attn output + ctx rows for selected queries
    scores_softmax_ctx<<<BB * HH * NTOP, 256, 0, stream>>>(Qp, Kp, Vp, topIdx, attn, ctx);

    // 6) output projection from ctx layout (WMMA bf16 GEMM)
    proj_wmma<<<pgrid, 256, 0, stream>>>(ctx, Wo, bo, out, 1);
}
